// OurModel_58360015618369
// MI455X (gfx1250) — matrix operand order repaired and checked by reading, not compiled, not run on hardware
//
#include <hip/hip_runtime.h>
#include <hip/hip_bf16.h>

typedef __attribute__((ext_vector_type(16))) _Float16 v16h;
typedef __attribute__((ext_vector_type(8)))  _Float16 v8h;
typedef __attribute__((ext_vector_type(8)))  float    v8f;
typedef __attribute__((ext_vector_type(4)))  float    v4f;

#define NB 4          // batch
#define PLANE 1024    // hb*wb plane length (contiguous); also number of p positions
#define STP 20        // staging pitch in floats (16B aligned, bank-skewed)
#define STW (32*STP)  // staging floats per wave

// ---------------------------------------------------------------------------
// Fused separable-4D-conv block. One workgroup (256 threads = 8 waves) handles
// one (b, p) site: GEMM  D[16 x 1024] = Wc[16 x K2] * im2col[K2 x 1024],
// K2 = 2*Cin*9 (branch1 p-neighbor planes ++ branch2 q-shifted center planes).
// im2col staged TRANSPOSED in LDS (fragment = two aligned ds_load_b128); all
// masking done at staging. Prev-layer GroupNorm+ReLU applied at load.
// Epilogue: per-wave LDS transpose -> uniform loop over COUT rows with fully
// coalesced 128B global stores (no divergence). Raw conv out (fp32) stored;
// per-sample sum/sumsq accumulated for GN.
// ---------------------------------------------------------------------------
template<int CIN, int COUT, bool ACT, int CHUNK>
__global__ __launch_bounds__(256)
void conv4d_wmma_kernel(const float* __restrict__ src,      // [NB, CIN, 1024, 1024] raw pre-GN
                        const float* __restrict__ w1,       // [COUT, CIN, 3, 3]
                        const float* __restrict__ b1,       // [COUT]
                        const float* __restrict__ w2,       // [COUT, CIN, 3, 3]
                        const float* __restrict__ b2,       // [COUT]
                        const float* __restrict__ gammaPrev,// [CIN] or null (layer 0)
                        const float* __restrict__ betaPrev, // [CIN] or null
                        const float* __restrict__ mrPrev,   // [NB][2] {mu, rstd} or null
                        float* __restrict__ dst,            // [NB, COUT, 1024, 1024]
                        float* __restrict__ statsAcc)       // [NB][2] {sum, sumsq}
{
    constexpr int K1     = CIN * 9;            // branch-1 rows
    constexpr int K2     = 2 * K1;             // total K
    constexpr int KSTEP  = (K2 + 31) / 32;     // 16x16x32 k-steps
    constexpr int ROWS   = KSTEP * 32;         // zero-padded K
    constexpr int KPAD   = ROWS + 8;           // im2col pitch (halves): 16B-aligned, bank-skewed
    constexpr int NCHUNK = PLANE / CHUNK;

    extern __shared__ char smem[];
    _Float16* Bmat  = (_Float16*)smem;                    // [CHUNK][KPAD] transposed im2col
    float*    stage = (float*)(Bmat + CHUNK * KPAD);      // [8 waves][32 cols][STP]
    _Float16* Wc    = (_Float16*)(stage + 8 * STW);       // [16][ROWS] combined weights
    float*    biasS = (float*)(Wc + 16 * ROWS);           // [16] b1+b2
    float*    gb    = biasS + 16;                         // [CIN][2] scale, shift
    float*    red   = gb + 2 * CIN;                       // [512] reduction scratch

    const int tid    = threadIdx.x;
    const int lane   = tid & 31;
    const int wv     = tid >> 5;
    const int bp     = blockIdx.x;              // b*1024 + p
    const int b      = bp >> 10;
    const int p      = bp & 1023;
    const int ha     = p >> 5, wa = p & 31;
    const int hiHalf = lane >> 4;
    const int ncol   = lane & 15;
    const float* srcB = src + (((size_t)b * CIN) << 20);
    float* dstB = dst + (((size_t)(b * COUT)) << 20) + ((size_t)p << 10);

    float mu = 0.f, rstd = 1.f;
    if (mrPrev) { mu = mrPrev[b * 2 + 0]; rstd = mrPrev[b * 2 + 1]; }
    if (tid < CIN) {
        float g  = gammaPrev ? gammaPrev[tid] : 1.f;
        float bt = betaPrev  ? betaPrev[tid]  : 0.f;
        gb[2 * tid + 0] = g * rstd;
        gb[2 * tid + 1] = bt - mu * rstd * g;
    }
    if (tid < 16) biasS[tid] = (tid < COUT) ? (b1[tid] + b2[tid]) : 0.f;
    // combined weight matrix Wc[m][k], zero padded
    for (int idx = tid; idx < 16 * ROWS; idx += 256) {
        int m = idx / ROWS;
        int k = idx - m * ROWS;
        float v = 0.f;
        if (m < COUT) {
            if (k < K1)      v = w1[m * K1 + k];
            else if (k < K2) v = w2[m * K1 + (k - K1)];
        }
        Wc[idx] = (_Float16)v;
    }
    __syncthreads();

    // ---- A fragments (contiguous K layout: lanes<16 hold K {0..15},
    //      lanes>=16 hold K {16..31}, rows m = lane%16) ----------------
    v16h afrag[KSTEP];
#pragma unroll
    for (int kk = 0; kk < KSTEP; ++kk) {
#pragma unroll
        for (int h = 0; h < 16; ++h) {
            int kl = h + hiHalf * 16;
            afrag[kk][h] = Wc[(lane & 15) * ROWS + kk * 32 + kl];
        }
    }
    // bias in registers (row set for this lane half)
    float biasReg[8];
#pragma unroll
    for (int j = 0; j < 8; ++j) biasReg[j] = biasS[j + hiHalf * 8];

    float s1 = 0.f, s2 = 0.f;

    for (int c = 0; c < NCHUNK; ++c) {
        const int qBase = c * CHUNK;
        if (c) __syncthreads();   // previous chunk's compute done before refill

        // ---- stage transposed im2col for this q-chunk ----------------------
        // k uniform per wave-iteration (CHUNK is a multiple of 256).
        for (int idx = tid; idx < ROWS * CHUNK; idx += 256) {
            int k  = idx / CHUNK;
            int ql = idx - k * CHUNK;
            int q  = qBase + ql;
            _Float16 hv = (_Float16)0.f;
            if (k < K1) {                       // branch 1: p-neighbor plane, same q
                int ci = k / 9, t = k - ci * 9;
                int nha = ha + t / 3 - 1, nwa = wa + (t % 3) - 1;
                if ((unsigned)nha < 32u && (unsigned)nwa < 32u) {  // uniform branch
                    float v = srcB[((size_t)ci << 20) + ((size_t)(nha * 32 + nwa) << 10) + q];
                    v = fmaf(v, gb[2 * ci], gb[2 * ci + 1]);
                    if (ACT) v = fmaxf(v, 0.f);
                    hv = (_Float16)v;
                }
            } else if (k < K2) {                // branch 2: center plane, shifted q
                int k2 = k - K1;
                int ci = k2 / 9, t = k2 - ci * 9;
                int dy = t / 3 - 1, dx = (t % 3) - 1;
                int hb = q >> 5, wb = q & 31;
                bool ok = ((unsigned)(hb + dy) < 32u) & ((unsigned)(wb + dx) < 32u);
                int qq = ok ? (q + dy * 32 + dx) : q;       // branchless clamp
                float v = srcB[((size_t)ci << 20) + ((size_t)p << 10) + qq];
                v = fmaf(v, gb[2 * ci], gb[2 * ci + 1]);
                if (ACT) v = fmaxf(v, 0.f);
                hv = ok ? (_Float16)v : (_Float16)0.f;
            }
            Bmat[ql * KPAD + k] = hv;
        }
        __syncthreads();

        // ---- compute: tile PAIRS (32 columns), round-robin over 8 waves ----
        for (int qp = wv; qp < CHUNK / 32; qp += 8) {
            const _Float16* br0 = Bmat + (qp * 32 + ncol) * KPAD;
            const _Float16* br1 = br0 + 16 * KPAD;
            v8f acc0 = {}, acc1 = {};
#pragma unroll
            for (int kk = 0; kk < KSTEP; ++kk) {
                union { v16h v; v8h h[2]; } bf0, bf1;
                bf0.h[0] = *(const v8h*)(br0 + kk * 32 + hiHalf * 16);
                bf0.h[1] = *(const v8h*)(br0 + kk * 32 + hiHalf * 16 + 8);
                bf1.h[0] = *(const v8h*)(br1 + kk * 32 + hiHalf * 16);
                bf1.h[1] = *(const v8h*)(br1 + kk * 32 + hiHalf * 16 + 8);
                acc0 = __builtin_amdgcn_wmma_f32_16x16x32_f16(
                    false, afrag[kk], false, bf0.v, (short)0, acc0, false, false);
                acc1 = __builtin_amdgcn_wmma_f32_16x16x32_f16(
                    false, afrag[kk], false, bf1.v, (short)0, acc1, false, false);
            }
            // bias in registers, then wave-private LDS transpose [col][m]
            union { v8f v; v4f q[2]; } a0, a1;
#pragma unroll
            for (int j = 0; j < 8; ++j) { a0.v[j] = acc0[j] + biasReg[j]; a1.v[j] = acc1[j] + biasReg[j]; }
            float* stw = stage + wv * STW;
            v4f* st0 = (v4f*)(stw + ncol * STP + hiHalf * 8);
            v4f* st1 = (v4f*)(stw + (16 + ncol) * STP + hiHalf * 8);
            st0[0] = a0.q[0]; st0[1] = a0.q[1];
            st1[0] = a1.q[0]; st1[1] = a1.q[1];
            // uniform row loop: coalesced 128B stores, stats on the fly
            const float* strow = stw + lane * STP;
            float* drow = dstB + qBase + qp * 32 + lane;
#pragma unroll
            for (int m = 0; m < COUT; ++m) {
                float v = strow[m];
                s1 += v; s2 += v * v;
                drow[(size_t)m << 20] = v;
            }
        }
    }

    // ---- workgroup reduction, one atomic per WG per stat -------------------
    red[tid] = s1; red[256 + tid] = s2;
    __syncthreads();
    for (int off = 128; off > 0; off >>= 1) {
        if (tid < off) { red[tid] += red[tid + off]; red[256 + tid] += red[256 + tid + off]; }
        __syncthreads();
    }
    if (tid == 0) {
        atomicAdd(&statsAcc[b * 2 + 0], red[0]);
        atomicAdd(&statsAcc[b * 2 + 1], red[256]);
    }
}

__global__ void init_stats_kernel(float* acc0) {
    if (threadIdx.x < 2 * NB) acc0[threadIdx.x] = 0.f;
}

__global__ void stats_kernel(const float* __restrict__ acc, float* __restrict__ mr,
                             float invCnt, float* __restrict__ nextAcc) {
    int t = threadIdx.x;
    if (t < NB) {
        float s1 = acc[t * 2 + 0], s2 = acc[t * 2 + 1];
        float m  = s1 * invCnt;
        float var = fmaxf(s2 * invCnt - m * m, 0.f);
        mr[t * 2 + 0] = m;
        mr[t * 2 + 1] = rsqrtf(var + 1e-5f);
    }
    if (nextAcc && t < 2 * NB) nextAcc[t] = 0.f;
}

__global__ void finalize_kernel(const float* __restrict__ src, const float* __restrict__ mr,
                                const float* __restrict__ gamma, const float* __restrict__ beta,
                                float* __restrict__ out, int n) {
    int i = blockIdx.x * blockDim.x + threadIdx.x;
    if (i < n) {
        int b = i >> 20;   // COUT=1 -> 1M elements per sample
        float v = (src[i] - mr[b * 2]) * mr[b * 2 + 1] * gamma[0] + beta[0];
        out[i] = fmaxf(v, 0.f);
    }
}

static size_t shbytes(int cin, int chunk) {
    int k2 = 2 * cin * 9;
    int kstep = (k2 + 31) / 32;
    int rows = kstep * 32;
    int kpad = rows + 8;
    return (size_t)chunk * kpad * 2   // Bmat
         + (size_t)8 * STW * 4        // stage
         + (size_t)16 * rows * 2      // Wc
         + 16 * 4                     // biasS
         + (size_t)2 * cin * 4        // gb
         + 512 * 4;                   // reduction
}

extern "C" void kernel_launch(void* const* d_in, const int* in_sizes, int n_in,
                              void* d_out, int out_size, void* d_ws, size_t ws_size,
                              hipStream_t stream) {
    (void)in_sizes; (void)n_in; (void)out_size; (void)ws_size;

    const float* corr = (const float*)d_in[0];
    auto P = [&](int L, int j) { return (const float*)d_in[1 + L * 6 + j]; };
    // per-layer: j = 0:w1, 1:b1, 2:w2, 3:b2, 4:gamma, 5:beta

    const size_t BUF = (size_t)NB * 10 * 1024 * 1024; // max feature elements (C=10)
    float* buf0     = (float*)d_ws;
    float* buf1     = buf0 + BUF;
    float* statsAcc = buf1 + BUF;        // 6 layers * [NB][2]
    float* statsMR  = statsAcc + 6 * 2 * NB;

    dim3 grid(NB * 1024), blk(256);
    const float inv10 = 1.f / (10.f * 1048576.f);
    const float inv1  = 1.f / (1048576.f);

    init_stats_kernel<<<1, 32, 0, stream>>>(statsAcc);

    // L0: 2 -> 10, raw corr input (identity GN, no ReLU)
    conv4d_wmma_kernel<2, 10, false, 1024><<<grid, blk, shbytes(2, 1024), stream>>>(
        corr, P(0,0), P(0,1), P(0,2), P(0,3), nullptr, nullptr, nullptr,
        buf0, statsAcc + 0);
    stats_kernel<<<1, 32, 0, stream>>>(statsAcc + 0, statsMR + 0, inv10, statsAcc + 8);

    // L1: 10 -> 10  (fuses GN0+ReLU at load)
    conv4d_wmma_kernel<10, 10, true, 512><<<grid, blk, shbytes(10, 512), stream>>>(
        buf0, P(1,0), P(1,1), P(1,2), P(1,3), P(0,4), P(0,5), statsMR + 0,
        buf1, statsAcc + 8);
    stats_kernel<<<1, 32, 0, stream>>>(statsAcc + 8, statsMR + 8, inv10, statsAcc + 16);

    // L2: 10 -> 1
    conv4d_wmma_kernel<10, 1, true, 512><<<grid, blk, shbytes(10, 512), stream>>>(
        buf1, P(2,0), P(2,1), P(2,2), P(2,3), P(1,4), P(1,5), statsMR + 8,
        buf0, statsAcc + 16);
    stats_kernel<<<1, 32, 0, stream>>>(statsAcc + 16, statsMR + 16, inv1, statsAcc + 24);

    // L3: 1 -> 10
    conv4d_wmma_kernel<1, 10, true, 1024><<<grid, blk, shbytes(1, 1024), stream>>>(
        buf0, P(3,0), P(3,1), P(3,2), P(3,3), P(2,4), P(2,5), statsMR + 16,
        buf1, statsAcc + 24);
    stats_kernel<<<1, 32, 0, stream>>>(statsAcc + 24, statsMR + 24, inv10, statsAcc + 32);

    // L4: 10 -> 10
    conv4d_wmma_kernel<10, 10, true, 512><<<grid, blk, shbytes(10, 512), stream>>>(
        buf1, P(4,0), P(4,1), P(4,2), P(4,3), P(3,4), P(3,5), statsMR + 24,
        buf0, statsAcc + 32);
    stats_kernel<<<1, 32, 0, stream>>>(statsAcc + 32, statsMR + 32, inv10, statsAcc + 40);

    // L5: 10 -> 1
    conv4d_wmma_kernel<10, 1, true, 512><<<grid, blk, shbytes(10, 512), stream>>>(
        buf0, P(5,0), P(5,1), P(5,2), P(5,3), P(4,4), P(4,5), statsMR + 32,
        buf1, statsAcc + 40);
    stats_kernel<<<1, 32, 0, stream>>>(statsAcc + 40, statsMR + 40, inv1, nullptr);

    // final GN + ReLU -> d_out  (4 * 1 * 32^4 fp32 elements)
    const int n = NB * 1024 * 1024;
    finalize_kernel<<<(n + 255) / 256, 256, 0, stream>>>(
        buf1, statsMR + 40, P(5,4), P(5,5), (float*)d_out, n);
}
